// PossTransformerLSTM_16054587752959
// MI455X (gfx1250) — matrix operand order repaired and checked by reading, not compiled, not run on hardware
//
#include <hip/hip_runtime.h>
#include <hip/hip_bf16.h>

typedef __attribute__((ext_vector_type(16))) __bf16          v16bf;
typedef __attribute__((ext_vector_type(16))) unsigned short  v16us;
typedef __attribute__((ext_vector_type(8)))  float           v8f;

__device__ __forceinline__ unsigned short f2bf(float f) {
  unsigned int u = __float_as_uint(f);
  u += 0x7FFFu + ((u >> 16) & 1u);          // round-to-nearest-even
  return (unsigned short)(u >> 16);
}
__device__ __forceinline__ unsigned int pack2(float a, float b) {
  return (unsigned int)f2bf(a) | ((unsigned int)f2bf(b) << 16);
}

__device__ __forceinline__ v8f wmma_bf16(v16us a, v16us b, v8f c) {
  union { v16us u; v16bf h; } A, B;
  A.u = a; B.u = b;
  return __builtin_amdgcn_wmma_f32_16x16x32_bf16(false, A.h, false, B.h,
                                                 (short)0, c, false, false);
}

// Fragment-order index helpers (one 32-deep K chunk = 512 bf16 = [lane][e]).
// A-matrix (16xK, M x K): lane = m + 16*hf, hf = kl>>4, e = kl&15
__device__ __forceinline__ int afrag_idx(int m, int kl) {
  int hf = kl >> 4;
  int e  = kl & 15;
  return (m + (hf << 4)) * 16 + e;
}
// B-matrix (Kx16, K x N): lane = nl + 16*hf, hf = kl>>4, e = kl&15
__device__ __forceinline__ int bfrag_idx(int nl, int kl) {
  int hf = kl >> 4;
  int e  = kl & 15;
  return (nl + (hf << 4)) * 16 + e;
}
__device__ __forceinline__ v16us frag_load(const unsigned short* p) {
  return *(const v16us*)p;     // 2x b128 loads, lane-contiguous
}

// ---------------- set-transformer encodings + positional encoding ----------
__global__ __launch_bounds__(256) void set_kernel(
    const float* __restrict__ coords, const float* __restrict__ w1,
    const float* __restrict__ b1, const float* __restrict__ w2,
    const float* __restrict__ b2, float* __restrict__ enc_in,
    float* __restrict__ hbuf) {
  __shared__ float sh[512];
  const int t = blockIdx.x, j = threadIdx.x;
  const int spos = t & 255;
  float w1c[4];
#pragma unroll
  for (int k = 0; k < 4; ++k) w1c[k] = w1[k * 256 + j];
  const float bb = b1[j];
  for (int team = 0; team < 2; ++team) {
    float mh = 0.f;
    for (int p = 0; p < 11; ++p) {
      const float* cp = coords + (long)t * 88 + (team * 11 + p) * 4;
      float s = bb + cp[0] * w1c[0] + cp[1] * w1c[1] + cp[2] * w1c[2] + cp[3] * w1c[3];
      mh += s > 0.f ? s : 0.f;
    }
    sh[team * 256 + j] = mh * (1.f / 11.f);
  }
  __syncthreads();
  float e1 = b2[j], e2 = b2[j];
  for (int k = 0; k < 256; ++k) {
    float w = w2[k * 256 + j];
    e1 += sh[k] * w;
    e2 += sh[256 + k] * w;
  }
  enc_in[(long)t * 512 + j] = e1;
  enc_in[(long)t * 512 + 256 + j] = e2;
  const float c0 = -9.210340371976184f / 512.f;  // -ln(10000)/512
  int d1 = j, d2 = 256 + j;
  float a1 = (float)spos * __expf((float)(d1 & ~1) * c0);
  float a2 = (float)spos * __expf((float)(d2 & ~1) * c0);
  float p1 = (d1 & 1) ? __cosf(a1) : __sinf(a1);
  float p2 = (d2 & 1) ? __cosf(a2) : __sinf(a2);
  hbuf[(long)t * 512 + j] = e1 + p1;
  hbuf[(long)t * 512 + 256 + j] = e2 + p2;
}

// ---------------- generic bf16-WMMA GEMM: C = A*B[:,bcol:] + bias ----------
// block = 4 waves, tile 16(M) x 64(N); K multiple of 32
__global__ __launch_bounds__(128) void gemm_kernel(
    const float* __restrict__ A, int lda, const float* __restrict__ B, int ldb,
    int bcol, const float* __restrict__ bias, float* __restrict__ C, int ldc,
    int ccol, int K, int relu) {
  __shared__ alignas(32) unsigned short AsF[512];    // A frag-order: [lane][16]
  __shared__ alignas(32) unsigned short BsF[2048];   // 4 waves x [lane][16]
  const int tid = threadIdx.x;
  const int wave = tid >> 5, lane = tid & 31;
  const int hf = lane >> 4;
  const int m0 = blockIdx.y << 4;
  const int n0 = blockIdx.x << 6;
  v8f acc = {};
  const int nk = K >> 5;
  for (int kc = 0; kc < nk; ++kc) {
    const int kb = kc << 5;
    // A tile 16x32 -> frag order, 2 bf16 per dword store
    for (int i = tid; i < 256; i += 128) {
      int m = i >> 4, kp = (i & 15) << 1;
      const float* ap = A + (long)(m0 + m) * lda + kb + kp;
      *(unsigned int*)&AsF[afrag_idx(m, kp)] = pack2(ap[0], ap[1]);
    }
    // B tile 32x64 -> frag order (pairs along k share a dword)
    for (int i = tid; i < 1024; i += 128) {
      int n = i & 63, kp = (i >> 6) << 1;
      const float* bp = B + (long)(kb + kp) * ldb + bcol + n0 + n;
      int idx = ((n >> 4) << 9) + bfrag_idx(n & 15, kp);
      *(unsigned int*)&BsF[idx] = pack2(bp[0], bp[ldb]);
    }
    __syncthreads();
    v16us af  = frag_load(AsF + lane * 16);
    v16us bfr = frag_load(BsF + ((wave << 5) + lane) * 16);
    acc = wmma_bf16(af, bfr, acc);
    __syncthreads();
  }
  const int row = lane & 15;
  const int nn = n0 + (wave << 4) + row;
  const float bv = bias[bcol + nn];
#pragma unroll
  for (int r = 0; r < 8; ++r) {
    int m = m0 + r + (hf << 3);
    float v = acc[r] + bv;
    if (relu) v = v > 0.f ? v : 0.f;
    C[(long)m * ldc + ccol + nn] = v;
  }
}

// ---------------- fused attention for one (b, h, 16-query block) -----------
__global__ __launch_bounds__(256) void attn_kernel(const float* __restrict__ qkv,
                                                   float* __restrict__ out,
                                                   int causal) {
  extern __shared__ char smem[];
  unsigned short* KtF = (unsigned short*)smem;            // 16 nt x 2 kc x 512
  unsigned short* VtF = KtF + 16384;                      // 4 nt x 8 kc x 512
  unsigned short* QtF = VtF + 16384;                      // 2 kc x 512
  float* Sc = (float*)(QtF + 1024);                       // 16x256 f32
  unsigned short* PbF = (unsigned short*)(Sc + 16 * 256); // 8 kc x 512
  float* red = (float*)(PbF + 4096);                      // 256
  float* rowmax = red + 256;                              // 16
  float* rowsum = rowmax + 16;                            // 16
  const int tid = threadIdx.x;
  const int wave = tid >> 5, lane = tid & 31;
  const int hf = lane >> 4, row = lane & 15;
  const int qb = blockIdx.x, h = blockIdx.y, b = blockIdx.z;
  const long tq0 = (long)b * 256 + qb * 16;
  // Q 16x64 (A-matrix): frag order
  for (int i = tid; i < 512; i += 256) {
    int m = i >> 5, dp = (i & 31) << 1;
    const float* qp = qkv + (tq0 + m) * 1536 + h * 64 + dp;
    int idx = ((dp >> 5) << 9) + afrag_idx(m, dp & 31);
    *(unsigned int*)&QtF[idx] = pack2(qp[0], qp[1]);
  }
  // K 256x64, used as B = K^T (kdim = d, n = key j): frag order
  for (int i = tid; i < 8192; i += 256) {
    int j = i >> 5, dp = (i & 31) << 1;
    const float* kp = qkv + ((long)b * 256 + j) * 1536 + 512 + h * 64 + dp;
    int idx = (((j >> 4) * 2 + (dp >> 5)) << 9) + bfrag_idx(j & 15, dp & 31);
    *(unsigned int*)&KtF[idx] = pack2(kp[0], kp[1]);
  }
  // V 256x64 (B-matrix, kdim = key j, n = d): frag order, pairs along j
  for (int i = tid; i < 8192; i += 256) {
    int jp = (i >> 6) << 1, d = i & 63;
    const float* vp = qkv + ((long)b * 256 + jp) * 1536 + 1024 + h * 64 + d;
    int idx = (((d >> 4) * 8 + (jp >> 5)) << 9) + bfrag_idx(d & 15, jp & 31);
    *(unsigned int*)&VtF[idx] = pack2(vp[0], vp[1536]);
  }
  __syncthreads();
  // scores = Q K^T / 8 (+ causal mask)
  for (int nt = wave; nt < 16; nt += 8) {
    v8f acc = {};
    for (int kc = 0; kc < 2; ++kc) {
      v16us af  = frag_load(QtF + (kc << 9) + lane * 16);
      v16us bfr = frag_load(KtF + ((nt * 2 + kc) << 9) + lane * 16);
      acc = wmma_bf16(af, bfr, acc);
    }
    int j = nt * 16 + row;
#pragma unroll
    for (int r = 0; r < 8; ++r) {
      int m = r + (hf << 3);
      float v = acc[r] * 0.125f;
      if (causal && j > qb * 16 + m) v += -1e9f;
      Sc[m * 256 + j] = v;
    }
  }
  __syncthreads();
  // softmax: 16 threads per row
  {
    int r = tid >> 4, g = tid & 15;
    float mx = -3.4e38f;
    for (int c = g * 16; c < g * 16 + 16; ++c) mx = fmaxf(mx, Sc[r * 256 + c]);
    red[tid] = mx;
    __syncthreads();
    if (g == 0) {
      float m2 = red[tid];
      for (int x = 1; x < 16; ++x) m2 = fmaxf(m2, red[tid + x]);
      rowmax[r] = m2;
    }
    __syncthreads();
    float rm = rowmax[r], sm = 0.f;
    for (int c = g * 16; c < g * 16 + 16; ++c) {
      float e = __expf(Sc[r * 256 + c] - rm);
      Sc[r * 256 + c] = e;
      sm += e;
    }
    red[tid] = sm;
    __syncthreads();
    if (g == 0) {
      float s2 = 0.f;
      for (int x = 0; x < 16; ++x) s2 += red[tid + x];
      rowsum[r] = s2;
    }
    __syncthreads();
    // P (A-matrix 16x256) in frag order
    float inv = 1.f / rowsum[r];
    for (int c = g * 16; c < g * 16 + 16; c += 2) {
      int idx = ((c >> 5) << 9) + afrag_idx(r, c & 31);
      *(unsigned int*)&PbF[idx] =
          pack2(Sc[r * 256 + c] * inv, Sc[r * 256 + c + 1] * inv);
    }
  }
  __syncthreads();
  // O = P V  (16x64, waves 0..3, one 16x16 tile each)
  if (wave < 4) {
    v8f acc = {};
    for (int kc = 0; kc < 8; ++kc) {
      v16us af  = frag_load(PbF + (kc << 9) + lane * 16);
      v16us bfr = frag_load(VtF + ((wave * 8 + kc) << 9) + lane * 16);
      acc = wmma_bf16(af, bfr, acc);
    }
#pragma unroll
    for (int r = 0; r < 8; ++r) {
      int m = r + (hf << 3);
      out[(tq0 + m) * 512 + h * 64 + wave * 16 + row] = acc[r];
    }
  }
}

// ---------------- residual + LayerNorm over 512 ----------------------------
__global__ __launch_bounds__(128) void ln_kernel(
    const float* __restrict__ x, const float* __restrict__ add,
    const float* __restrict__ g, const float* __restrict__ bv,
    float* __restrict__ out) {
  __shared__ float red[128];
  const int t = blockIdx.x, tid = threadIdx.x;
  float v[4];
  float s = 0.f;
#pragma unroll
  for (int j = 0; j < 4; ++j) {
    int d = tid + (j << 7);
    v[j] = x[(long)t * 512 + d] + add[(long)t * 512 + d];
    s += v[j];
  }
  red[tid] = s;
  __syncthreads();
  for (int st = 64; st > 0; st >>= 1) {
    if (tid < st) red[tid] += red[tid + st];
    __syncthreads();
  }
  float mean = red[0] * (1.f / 512.f);
  __syncthreads();
  s = 0.f;
#pragma unroll
  for (int j = 0; j < 4; ++j) {
    float d0 = v[j] - mean;
    s += d0 * d0;
  }
  red[tid] = s;
  __syncthreads();
  for (int st = 64; st > 0; st >>= 1) {
    if (tid < st) red[tid] += red[tid + st];
    __syncthreads();
  }
  float rstd = rsqrtf(red[0] * (1.f / 512.f) + 1e-5f);
#pragma unroll
  for (int j = 0; j < 4; ++j) {
    int d = tid + (j << 7);
    out[(long)t * 512 + d] = (v[j] - mean) * rstd * g[d] + bv[d];
  }
}

// ---------------- ball head: [4096,512] @ [512,2] --------------------------
__global__ __launch_bounds__(64) void ball_kernel(
    const float* __restrict__ dec, const float* __restrict__ bw,
    const float* __restrict__ bb, float* __restrict__ ball) {
  __shared__ float r0[64], r1[64];
  int t = blockIdx.x, tid = threadIdx.x;
  float a0 = 0.f, a1 = 0.f;
  for (int d = tid; d < 512; d += 64) {
    float x = dec[(long)t * 512 + d];
    a0 += x * bw[d * 2 + 0];
    a1 += x * bw[d * 2 + 1];
  }
  r0[tid] = a0; r1[tid] = a1;
  __syncthreads();
  for (int st = 32; st > 0; st >>= 1) {
    if (tid < st) { r0[tid] += r0[tid + st]; r1[tid] += r1[tid + st]; }
    __syncthreads();
  }
  if (tid == 0) {
    ball[t * 2 + 0] = r0[0] + bb[0];
    ball[t * 2 + 1] = r1[0] + bb[1];
  }
}

// ------- LSTM static input rows: [ct(88) | et(512) | cd(1) | prev(2) | pad] -
__global__ __launch_bounds__(128) void xstatic_kernel(
    const float* __restrict__ coords, const float* __restrict__ enc_in,
    const float* __restrict__ ball, unsigned short* __restrict__ Xs) {
  int t = blockIdx.x;            // t = b*256 + s
  int b = t >> 8, sp = t & 255;
  unsigned short* dst = Xs + (long)(sp * 16 + b) * 608;
  for (int i = threadIdx.x; i < 608; i += 128) {
    float v;
    if (i < 88) v = coords[(long)t * 88 + i];
    else if (i < 600) v = enc_in[(long)t * 512 + (i - 88)];
    else if (i == 600) {
      float bx = ball[t * 2], by = ball[t * 2 + 1];
      float mind = 3.4e38f;
      for (int p = 0; p < 22; ++p) {
        float dx = bx - coords[(long)t * 88 + 2 * p];
        float dy = by - coords[(long)t * 88 + 2 * p + 1];
        mind = fminf(mind, sqrtf(dx * dx + dy * dy));
      }
      v = mind;
    } else v = 0.f;   // prev slots + pad (prev patched in LSTM kernel)
    dst[i] = f2bf(v);
  }
}

// -- pack [wih^T ; 0pad ; whh^T] bf16 in WMMA B-frag order -------------------
// layout: [nt(128)][kc(35)][lane(32)][e(16)]; n = nt*16+(lane&15),
// k = kc*32 + (lane>>4)*16 + e
__global__ __launch_bounds__(256) void wcat_kernel(
    const float* __restrict__ wih, const float* __restrict__ whh,
    unsigned short* __restrict__ Wc) {
  long idx = (long)blockIdx.x * blockDim.x + threadIdx.x;
  const long total = (long)128 * 35 * 512;
  for (; idx < total; idx += (long)gridDim.x * blockDim.x) {
    int within = (int)(idx & 511);
    int blk = (int)(idx >> 9);
    int kc = blk % 35, nt = blk / 35;
    int lane = within >> 4, e = within & 15;
    int n = nt * 16 + (lane & 15);
    int k = kc * 32 + ((lane >> 4) << 4) + e;
    float v = 0.f;
    if (k < 603) v = wih[(long)n * 603 + k];
    else if (k >= 608 && k < 1120) v = whh[(long)n * 512 + (k - 608)];
    Wc[idx] = f2bf(v);
  }
}

// ---------------- persistent LSTM: 1 block, 32 waves, 256 steps ------------
__global__ __launch_bounds__(1024) void lstm_kernel(
    const unsigned short* __restrict__ Xs, const unsigned short* __restrict__ Wc,
    const float* __restrict__ bih, const float* __restrict__ bhh,
    const float* __restrict__ sw, const float* __restrict__ sb,
    float* __restrict__ out) {
  extern __shared__ char smem[];
  unsigned short* AbfF = (unsigned short*)smem;             // 35 kc x 512 (frag)
  float* G   = (float*)(smem + 35840);                      // 16 x 2048
  float* cx  = (float*)(smem + 35840 + 131072);             // 16 x 512
  float* soL = (float*)(smem + 35840 + 131072 + 32768);     // 32
  float* red = soL + 32;                                    // 1024
  const int tid = threadIdx.x;
  const int wave = tid >> 5, lane = tid & 31;
  const int hf = lane >> 4, row = lane & 15;
  for (int i = tid; i < 16 * 512; i += 1024) {
    int b = i >> 9, j = i & 511;
    int k = 608 + j;
    cx[i] = 0.f;
    AbfF[((k >> 5) << 9) + afrag_idx(b, k & 31)] = 0;       // hx = 0
  }
  if (tid < 32) soL[tid] = 0.f;
  __syncthreads();
  for (int sp = 0; sp < 256; ++sp) {
    // phase 1: load static inputs (frag order), patch prev-output slots
    for (int i = tid; i < 16 * 608; i += 1024) {
      int b = i / 608, k = i - b * 608;
      unsigned short v = Xs[(long)(sp * 16 + b) * 608 + k];
      if (k == 601) v = f2bf(soL[b * 2 + 0]);
      else if (k == 602) v = f2bf(soL[b * 2 + 1]);
      AbfF[((k >> 5) << 9) + afrag_idx(b, k & 31)] = v;
    }
    __syncthreads();
    // phase 2: gates = A[16x1120] @ Wc[1120x2048] + biases
    {
      v8f acc[4] = {{}, {}, {}, {}};
      const int nt0 = wave << 2;
      for (int kc = 0; kc < 35; ++kc) {
        v16us af = frag_load(AbfF + (kc << 9) + lane * 16);
        if (kc + 1 < 35)
          __builtin_prefetch((const void*)(Wc + (((long)nt0 * 35 + kc + 1) << 9)), 0, 1);
#pragma unroll
        for (int tt = 0; tt < 4; ++tt) {
          v16us bfr = frag_load(Wc + (((long)(nt0 + tt) * 35 + kc) << 9) + lane * 16);
          acc[tt] = wmma_bf16(af, bfr, acc[tt]);
        }
      }
#pragma unroll
      for (int tt = 0; tt < 4; ++tt) {
        int n = (nt0 + tt) * 16 + row;
        float bsum = bih[n] + bhh[n];
#pragma unroll
        for (int r = 0; r < 8; ++r) {
          int m = r + (hf << 3);
          G[m * 2048 + n] = acc[tt][r] + bsum;
        }
      }
    }
    __syncthreads();
    // phase 3: LSTM cell elementwise
    for (int i = tid; i < 16 * 512; i += 1024) {
      int b = i >> 9, j = i & 511;
      float ii = G[b * 2048 + j];
      float ff = G[b * 2048 + 512 + j];
      float gg = G[b * 2048 + 1024 + j];
      float oo = G[b * 2048 + 1536 + j];
      float si = 1.f / (1.f + __expf(-ii));
      float sf = 1.f / (1.f + __expf(-ff));
      float so = 1.f / (1.f + __expf(-oo));
      float cn = sf * cx[i] + si * tanhf(gg);
      float hn = so * tanhf(cn);
      cx[i] = cn;
      int k = 608 + j;
      AbfF[((k >> 5) << 9) + afrag_idx(b, k & 31)] = f2bf(hn);  // hx for next step
      G[b * 2048 + j] = hn;                                     // f32 hx for head
    }
    __syncthreads();
    // phase 4: so = (hn @ step_w + step_b) * pitch
    {
      int pair = tid >> 5;                  // 32 pairs (b, c)
      int b = pair >> 1, c = pair & 1, l = tid & 31;
      float p = 0.f;
      for (int j = l; j < 512; j += 32) p += G[b * 2048 + j] * sw[j * 2 + c];
      red[tid] = p;
      __syncthreads();
      if (l == 0) {
        float s2 = 0.f;
        for (int x = 0; x < 32; ++x) s2 += red[pair * 32 + x];
        float v = (s2 + sb[c]) * (c == 0 ? 105.f : 68.f);
        soL[pair] = v;
        out[((long)b * 256 + sp) * 2 + c] = v;
      }
    }
    __syncthreads();
  }
}

extern "C" void kernel_launch(void* const* d_in, const int* in_sizes, int n_in,
                              void* d_out, int out_size, void* d_ws, size_t ws_size,
                              hipStream_t stream) {
  (void)in_sizes; (void)n_in; (void)out_size; (void)ws_size;
  const float* coords   = (const float*)d_in[1];
  const float* set_w1   = (const float*)d_in[2];
  const float* set_b1   = (const float*)d_in[3];
  const float* set_w2   = (const float*)d_in[4];
  const float* set_b2   = (const float*)d_in[5];
  const float* e_qkv_w  = (const float*)d_in[6];
  const float* e_qkv_b  = (const float*)d_in[7];
  const float* e_out_w  = (const float*)d_in[8];
  const float* e_out_b  = (const float*)d_in[9];
  const float* ds_qkv_w = (const float*)d_in[10];
  const float* ds_qkv_b = (const float*)d_in[11];
  const float* ds_out_w = (const float*)d_in[12];
  const float* ds_out_b = (const float*)d_in[13];
  const float* dc_qkv_w = (const float*)d_in[14];
  const float* dc_qkv_b = (const float*)d_in[15];
  const float* dc_out_w = (const float*)d_in[16];
  const float* dc_out_b = (const float*)d_in[17];
  const float* e_ff1_w  = (const float*)d_in[18];
  const float* e_ff1_b  = (const float*)d_in[19];
  const float* e_ff2_w  = (const float*)d_in[20];
  const float* e_ff2_b  = (const float*)d_in[21];
  const float* d_ff1_w  = (const float*)d_in[22];
  const float* d_ff1_b  = (const float*)d_in[23];
  const float* d_ff2_w  = (const float*)d_in[24];
  const float* d_ff2_b  = (const float*)d_in[25];
  const float* eln1g = (const float*)d_in[26];
  const float* eln1b = (const float*)d_in[27];
  const float* eln2g = (const float*)d_in[28];
  const float* eln2b = (const float*)d_in[29];
  const float* dln1g = (const float*)d_in[30];
  const float* dln1b = (const float*)d_in[31];
  const float* dln2g = (const float*)d_in[32];
  const float* dln2b = (const float*)d_in[33];
  const float* dln3g = (const float*)d_in[34];
  const float* dln3b = (const float*)d_in[35];
  const float* lstm_wih = (const float*)d_in[36];
  const float* lstm_whh = (const float*)d_in[37];
  const float* lstm_bih = (const float*)d_in[38];
  const float* lstm_bhh = (const float*)d_in[39];
  const float* ball_w = (const float*)d_in[42];
  const float* ball_b = (const float*)d_in[43];
  const float* step_w = (const float*)d_in[44];
  const float* step_b = (const float*)d_in[45];

  char* ws = (char*)d_ws;
  size_t o = 0;
  auto alloc = [&](size_t bytes) -> char* {
    char* p = ws + o; o += (bytes + 255) & ~(size_t)255; return p;
  };
  float* enc_in  = (float*)alloc((size_t)4096 * 512 * 4);
  float* hbuf    = (float*)alloc((size_t)4096 * 512 * 4);
  float* membuf  = (float*)alloc((size_t)4096 * 512 * 4);
  float* decbuf  = (float*)alloc((size_t)4096 * 512 * 4);
  float* qkvbuf  = (float*)alloc((size_t)4096 * 1536 * 4);
  float* attnbuf = (float*)alloc((size_t)4096 * 512 * 4);
  float* projbuf = (float*)alloc((size_t)4096 * 512 * 4);
  float* ffbuf   = (float*)alloc((size_t)4096 * 2048 * 4);
  float* ballbuf = (float*)alloc((size_t)4096 * 2 * 4);
  unsigned short* Xs = (unsigned short*)alloc((size_t)256 * 16 * 608 * 2);
  unsigned short* Wc = (unsigned short*)alloc((size_t)128 * 35 * 512 * 2);

  const size_t attn_lds = 16384 * 2 + 16384 * 2 + 1024 * 2 + 16 * 256 * 4 +
                          4096 * 2 + (256 + 32) * 4;

  set_kernel<<<dim3(4096), dim3(256), 0, stream>>>(coords, set_w1, set_b1,
                                                   set_w2, set_b2, enc_in, hbuf);
  // -------- encoder --------
  for (int i = 0; i < 6; ++i) {
    const float* src = (i == 0) ? hbuf : membuf;
    gemm_kernel<<<dim3(24, 256), dim3(128), 0, stream>>>(
        src, 512, e_qkv_w + (size_t)i * 512 * 1536, 1536, 0,
        e_qkv_b + (size_t)i * 1536, qkvbuf, 1536, 0, 512, 0);
    attn_kernel<<<dim3(16, 8, 16), dim3(256), attn_lds, stream>>>(qkvbuf, attnbuf, 0);
    gemm_kernel<<<dim3(8, 256), dim3(128), 0, stream>>>(
        attnbuf, 512, e_out_w + (size_t)i * 512 * 512, 512, 0,
        e_out_b + (size_t)i * 512, projbuf, 512, 0, 512, 0);
    ln_kernel<<<dim3(4096), dim3(128), 0, stream>>>(src, projbuf, eln1g + i * 512,
                                                    eln1b + i * 512, membuf);
    gemm_kernel<<<dim3(32, 256), dim3(128), 0, stream>>>(
        membuf, 512, e_ff1_w + (size_t)i * 512 * 2048, 2048, 0,
        e_ff1_b + (size_t)i * 2048, ffbuf, 2048, 0, 512, 1);
    gemm_kernel<<<dim3(8, 256), dim3(128), 0, stream>>>(
        ffbuf, 2048, e_ff2_w + (size_t)i * 2048 * 512, 512, 0,
        e_ff2_b + (size_t)i * 512, projbuf, 512, 0, 2048, 0);
    ln_kernel<<<dim3(4096), dim3(128), 0, stream>>>(membuf, projbuf, eln2g + i * 512,
                                                    eln2b + i * 512, membuf);
  }
  // -------- decoder --------
  for (int i = 0; i < 6; ++i) {
    const float* src = (i == 0) ? hbuf : decbuf;
    gemm_kernel<<<dim3(24, 256), dim3(128), 0, stream>>>(
        src, 512, ds_qkv_w + (size_t)i * 512 * 1536, 1536, 0,
        ds_qkv_b + (size_t)i * 1536, qkvbuf, 1536, 0, 512, 0);
    attn_kernel<<<dim3(16, 8, 16), dim3(256), attn_lds, stream>>>(qkvbuf, attnbuf, 1);
    gemm_kernel<<<dim3(8, 256), dim3(128), 0, stream>>>(
        attnbuf, 512, ds_out_w + (size_t)i * 512 * 512, 512, 0,
        ds_out_b + (size_t)i * 512, projbuf, 512, 0, 512, 0);
    ln_kernel<<<dim3(4096), dim3(128), 0, stream>>>(src, projbuf, dln1g + i * 512,
                                                    dln1b + i * 512, decbuf);
    // cross: q from dec, k/v from encoder memory
    gemm_kernel<<<dim3(8, 256), dim3(128), 0, stream>>>(
        decbuf, 512, dc_qkv_w + (size_t)i * 512 * 1536, 1536, 0,
        dc_qkv_b + (size_t)i * 1536, qkvbuf, 1536, 0, 512, 0);
    gemm_kernel<<<dim3(16, 256), dim3(128), 0, stream>>>(
        membuf, 512, dc_qkv_w + (size_t)i * 512 * 1536, 1536, 512,
        dc_qkv_b + (size_t)i * 1536, qkvbuf, 1536, 512, 512, 0);
    attn_kernel<<<dim3(16, 8, 16), dim3(256), attn_lds, stream>>>(qkvbuf, attnbuf, 0);
    gemm_kernel<<<dim3(8, 256), dim3(128), 0, stream>>>(
        attnbuf, 512, dc_out_w + (size_t)i * 512 * 512, 512, 0,
        dc_out_b + (size_t)i * 512, projbuf, 512, 0, 512, 0);
    ln_kernel<<<dim3(4096), dim3(128), 0, stream>>>(decbuf, projbuf, dln2g + i * 512,
                                                    dln2b + i * 512, decbuf);
    gemm_kernel<<<dim3(32, 256), dim3(128), 0, stream>>>(
        decbuf, 512, d_ff1_w + (size_t)i * 512 * 2048, 2048, 0,
        d_ff1_b + (size_t)i * 2048, ffbuf, 2048, 0, 512, 1);
    gemm_kernel<<<dim3(8, 256), dim3(128), 0, stream>>>(
        ffbuf, 2048, d_ff2_w + (size_t)i * 2048 * 512, 512, 0,
        d_ff2_b + (size_t)i * 512, projbuf, 512, 0, 2048, 0);
    ln_kernel<<<dim3(4096), dim3(128), 0, stream>>>(decbuf, projbuf, dln3g + i * 512,
                                                    dln3b + i * 512, decbuf);
  }
  // -------- heads + LSTM --------
  ball_kernel<<<dim3(4096), dim3(64), 0, stream>>>(decbuf, ball_w, ball_b, ballbuf);
  xstatic_kernel<<<dim3(4096), dim3(128), 0, stream>>>(coords, enc_in, ballbuf, Xs);
  wcat_kernel<<<dim3(2048), dim3(256), 0, stream>>>(lstm_wih, lstm_whh, Wc);
  const size_t lstm_lds = 35840 + 131072 + 32768 + 128 + 4096;
  lstm_kernel<<<dim3(1), dim3(1024), lstm_lds, stream>>>(
      Xs, Wc, lstm_bih, lstm_bhh, step_w, step_b, (float*)d_out);
}
